// GraphConvTriDense_36129264894619
// MI455X (gfx1250) — matrix operand order repaired and checked by reading, not compiled, not run on hardware
//
#include <hip/hip_runtime.h>

// GraphConvTriDense on MI455X (gfx1250).
// Memory-roofline: ~1.07 GB of f32 adjacency traffic -> ~46us floor @ 23.3 TB/s.
// AI = 64 FLOP/byte on adjacency -> f32 WMMA (16x16x4) would be compute bound;
// bf16 WMMA (16x16x32, f32 accumulate) keeps us on the memory roof.
// A operand (raw adjacency): Tensor Data Mover DMA (tensor_load_to_lds) into
// double-buffered LDS (TENSORcnt-tracked, overlapped with WMMA), f32->bf16
// conversion fused into the fragment read. B operand: degree-prescaled bf16
// features, pre-swizzled in global memory to the WMMA B-fragment layout
// (L2-resident, two coalesced 16B loads per fragment). Row degree factor,
// self term and relu applied in the f32 epilogue.

#define NS   4096
#define NT   8192
#define DIN  256
#define DOUT 128

typedef __attribute__((ext_vector_type(16))) __bf16 v16bf;
typedef __attribute__((ext_vector_type(8)))  __bf16 v8bf;
typedef __attribute__((ext_vector_type(8)))  float  v8f;
typedef __attribute__((ext_vector_type(4)))  unsigned int u32x4;
typedef __attribute__((ext_vector_type(8)))  int i32x8;
typedef __attribute__((ext_vector_type(4)))  int i32x4;

union V16 { v16bf v; v8bf h[2]; };

#define AF_STRIDE 36                       // f32 LDS row stride (32 data + 4 pad dwords)
#define ABUF_ELEMS (32 * AF_STRIDE)        // 1152 f32 per buffer
#define ABUF_BYTES (ABUF_ELEMS * 4)        // 4608 B

// B-fragment swizzle: element (k,n) -> position so a wave's fragment is two
// contiguous 16B chunks per lane. lane = nn + ((kk&8)<<1); slot = (kk&7) | ((kk&16)>>1).
__device__ __forceinline__ size_t swz_idx(int k, int n) {
  int kt = k >> 5, kk = k & 31, nt = n >> 4, nn = n & 15;
  return ((size_t)((kt * 8 + nt) * 32 + (nn + ((kk & 8) << 1)))) * 16 + (size_t)((kk & 7) | ((kk & 16) >> 1));
}

// ---------------- Tensor Data Mover: 2D f32 tile -> LDS --------------------
// D# per CDNA5 ISA: group0 {count=1, lds_addr, global_addr,
// type=2}; group1 {data_size=4B, pad_enable, pad_interval=32dw, pad_amount=4dw,
// tensor_dim0=32, tensor_dim1=rows, tile_dim0=32, tile_dim1=rows,
// tensor_dim0_stride=lda}; groups 2/3 zero (2D tensor).

__device__ __forceinline__ void tdm_load_rows(const float* g, size_t lda,
                                              unsigned int lds_byte, int rows) {
  unsigned long long ga = (unsigned long long)(size_t)g;
  u32x4 g0 = { 1u,                                  // count=1, user descriptor
               lds_byte,                            // lds_addr (bytes)
               (unsigned int)ga,                    // global_addr[31:0]
               ((unsigned int)(ga >> 32) & 0x01FFFFFFu) | (2u << 30) };  // [56:32], type=2
  const unsigned int d0 = (2u << 16)   // data_size = 4 bytes
                        | (1u << 20)   // pad_enable
                        | (4u << 22)   // pad_interval: 32 dwords
                        | (3u << 25);  // pad_amount: 4 dwords -> 36-dword rows
  i32x8 g1 = { (int)d0,
               (int)(32u << 16),                    // tensor_dim0 = 32 (lo16 @ [63:48])
               (int)((unsigned int)rows << 16),     // tensor_dim1 lo16 @ [95:80]
               (int)(32u << 16),                    // tile_dim0 = 32 @ [127:112]
               (int)rows,                           // tile_dim1 @ [143:128]
               (int)(unsigned int)lda,              // tensor_dim0_stride[31:0]
               (int)((unsigned int)((unsigned long long)lda >> 32) & 0xFFFF),
               0 };
  i32x4 gz = { 0, 0, 0, 0 };
#if defined(__clang_major__) && (__clang_major__ >= 23)
  i32x8 gz8 = { 0, 0, 0, 0, 0, 0, 0, 0 };
  __builtin_amdgcn_tensor_load_to_lds(g0, g1, gz, gz, gz8, 0);
#else
  __builtin_amdgcn_tensor_load_to_lds(g0, g1, gz, gz, 0);
#endif
}

// Each wave DMA-loads its 8-row slice of the block's 32x32 f32 tile.
__device__ __forceinline__ void issue_slice(const float* A, size_t lda, int row0,
                                            int k0, unsigned int buf_byte, int wv) {
  tdm_load_rows(A + (size_t)(row0 + wv * 8) * lda + k0, lda,
                buf_byte + (unsigned int)(wv * 8 * AF_STRIDE * 4), 8);
}

// ---------------- degree kernels ----------------

__global__ void k_deg_init(float* __restrict__ deg) {
  deg[blockIdx.x * 256 + threadIdx.x] = 1.0f;  // the (+1) before sqrt
}

__global__ void k_row_deg(const float* __restrict__ adj, const float* __restrict__ adj_s,
                          float* __restrict__ rs) {
  const int i = blockIdx.x, tid = threadIdx.x;
  float s = 0.f;
  const float4* a = (const float4*)(adj + (size_t)i * NT);
  for (int t = tid; t < NT / 4; t += 256) { float4 v = a[t]; s += (v.x + v.y) + (v.z + v.w); }
  const float4* b = (const float4*)(adj_s + (size_t)i * NS);
  for (int t = tid; t < NS / 4; t += 256) { float4 v = b[t]; s += (v.x + v.y) + (v.z + v.w); }
  __shared__ float red[256];
  red[tid] = s;
  __syncthreads();
  for (int off = 128; off > 0; off >>= 1) {
    if (tid < off) red[tid] += red[tid + off];
    __syncthreads();
  }
  if (tid == 0) rs[i] = rsqrtf(red[0] + 1.0f);
}

__global__ void k_col_deg(const float* __restrict__ adj, const float* __restrict__ adj_t,
                          float* __restrict__ deg) {
  const int j  = blockIdx.x * 256 + threadIdx.x;
  const int by = blockIdx.y;
  float s = 0.f;
  for (int i = by * (NS / 8); i < (by + 1) * (NS / 8); ++i) s += adj[(size_t)i * NT + j];
  for (int i = by * (NT / 8); i < (by + 1) * (NT / 8); ++i) s += adj_t[(size_t)i * NT + j];
  atomicAdd(&deg[j], s);
}

__global__ void k_rt(const float* __restrict__ deg, float* __restrict__ rt) {
  const int j = blockIdx.x * 256 + threadIdx.x;
  rt[j] = rsqrtf(deg[j]);
}

// ---------------- projection: x = inp_s @ W, y = inp_t @ W (f32 VALU) ------
// Also emits the degree-prescaled bf16 B-operands in WMMA fragment layout.

__global__ void k_proj(const float* __restrict__ inp_s, const float* __restrict__ inp_t,
                       const float* __restrict__ W,
                       const float* __restrict__ rs, const float* __restrict__ rt,
                       float* __restrict__ x_f32, float* __restrict__ y_f32,
                       __bf16* __restrict__ xb, __bf16* __restrict__ yb) {
  const int tid = threadIdx.x;
  const int row = blockIdx.x * 8 + (tid >> 5);  // one row per wave
  const int cg  = tid & 31;                     // 32 groups of 4 cols = 128
  const float* inp;
  float* outf;
  __bf16* outb;
  float r;
  int lrow;
  if (row < NS) {
    lrow = row; inp = inp_s + (size_t)row * DIN; outf = x_f32; outb = xb; r = rs[row];
  } else {
    lrow = row - NS; inp = inp_t + (size_t)lrow * DIN; outf = y_f32; outb = yb; r = rt[lrow];
  }
  const float4* W4 = (const float4*)W;
  float4 acc = {0.f, 0.f, 0.f, 0.f};
  for (int k = 0; k < DIN; ++k) {
    const float a = inp[k];
    const float4 w = W4[k * (DOUT / 4) + cg];
    acc.x += a * w.x; acc.y += a * w.y; acc.z += a * w.z; acc.w += a * w.w;
  }
  const int n0 = cg * 4;
  *(float4*)(outf + (size_t)lrow * DOUT + n0) = acc;
  outb[swz_idx(lrow, n0 + 0)] = (__bf16)(acc.x * r);
  outb[swz_idx(lrow, n0 + 1)] = (__bf16)(acc.y * r);
  outb[swz_idx(lrow, n0 + 2)] = (__bf16)(acc.z * r);
  outb[swz_idx(lrow, n0 + 3)] = (__bf16)(acc.w * r);
}

// ---------------- WMMA inner steps ----------------

// B fragments from the pre-swizzled global operand + 4 WMMAs.
__device__ __forceinline__ void mma_b(const V16& a, const __bf16* __restrict__ Bsw,
                                      int kt, int wn, int lid, v8f acc[4]) {
#pragma unroll
  for (int nt = 0; nt < 4; ++nt) {
    const int ntAbs = wn * 4 + nt;
    const __bf16* bp = Bsw + ((size_t)(kt * 8 + ntAbs) * 32 + lid) * 16;
    V16 b;
    b.h[0] = *(const v8bf*)bp;
    b.h[1] = *(const v8bf*)(bp + 8);
    acc[nt] = __builtin_amdgcn_wmma_f32_16x16x32_bf16(false, a.v, false, b.v,
                                                      (short)0, acc[nt], false, false);
  }
}

// A fragment from f32 LDS tile (K-contiguous rows), cvt to bf16 in-register.
// 16-bit A layout: lane<16 holds K ko..ko+7 / 16+ko..; lanes>=16 ko=8.
__device__ __forceinline__ void mma_step_direct(const float* buf,
                                                const __bf16* __restrict__ Bsw,
                                                int kt, int wm, int wn, int lid,
                                                v8f acc[4]) {
  const int mrow = wm * 16 + (lid & 15);
  const int ko = (lid >> 4) * 8;
  const float* rp = buf + mrow * AF_STRIDE;
  const float4 a0 = *(const float4*)(rp + ko);
  const float4 a1 = *(const float4*)(rp + ko + 4);
  const float4 a2 = *(const float4*)(rp + 16 + ko);
  const float4 a3 = *(const float4*)(rp + 20 + ko);
  V16 a;
  a.v[0]  = (__bf16)a0.x; a.v[1]  = (__bf16)a0.y; a.v[2]  = (__bf16)a0.z; a.v[3]  = (__bf16)a0.w;
  a.v[4]  = (__bf16)a1.x; a.v[5]  = (__bf16)a1.y; a.v[6]  = (__bf16)a1.z; a.v[7]  = (__bf16)a1.w;
  a.v[8]  = (__bf16)a2.x; a.v[9]  = (__bf16)a2.y; a.v[10] = (__bf16)a2.z; a.v[11] = (__bf16)a2.w;
  a.v[12] = (__bf16)a3.x; a.v[13] = (__bf16)a3.y; a.v[14] = (__bf16)a3.z; a.v[15] = (__bf16)a3.w;
  mma_b(a, Bsw, kt, wn, lid, acc);
}

// A^T fragment: tile in LDS is [ii][jj] (un-transposed DMA); gather column jr.
__device__ __forceinline__ void mma_step_trans(const float* buf,
                                               const __bf16* __restrict__ Bsw,
                                               int kt, int wm, int wn, int lid,
                                               v8f acc[4]) {
  const int jr = wm * 16 + (lid & 15);
  const int ko = (lid >> 4) * 8;
  V16 a;
#pragma unroll
  for (int e = 0; e < 8; ++e) a.v[e]     = (__bf16)buf[(ko + e) * AF_STRIDE + jr];
#pragma unroll
  for (int e = 0; e < 8; ++e) a.v[8 + e] = (__bf16)buf[(16 + ko + e) * AF_STRIDE + jr];
  mma_b(a, Bsw, kt, wn, lid, acc);
}

// ---------------- x-update ----------------
// x_new = relu( rs[m] * ( x[m] + adj_s@(rs*x) + adj@(rt*y) ) ); also emit
// xnb = bf16(rs * x_new) pre-swizzled for the y-update cross term.

__global__ void __launch_bounds__(128) k_xupdate(const float* __restrict__ adj,
                                                 const float* __restrict__ adj_s,
                                                 const __bf16* __restrict__ xb,
                                                 const __bf16* __restrict__ yb,
                                                 const float* __restrict__ x_f32,
                                                 const float* __restrict__ rs,
                                                 float* __restrict__ out_x,
                                                 __bf16* __restrict__ xnb) {
  __shared__ float ldsA[2 * ABUF_ELEMS];
  const int tid = threadIdx.x;
  const int lid = tid & 31;
  const int wv = __builtin_amdgcn_readfirstlane(tid >> 5);
  const int wm = wv >> 1, wn = wv & 1;
  const int row0 = blockIdx.x * 32;
  const unsigned int lbase = (unsigned int)(size_t)(void*)&ldsA[0];
  v8f acc[4] = {};

  // phase 1: adj_s @ (rs*x), K = NS
  issue_slice(adj_s, NS, row0, 0, lbase, wv);
  for (int kt = 0; kt < NS / 32; ++kt) {
    const float* cur = ldsA + (kt & 1) * ABUF_ELEMS;
    if (kt + 1 < NS / 32) {
      issue_slice(adj_s, NS, row0, (kt + 1) * 32,
                  lbase + (unsigned int)(((kt + 1) & 1) * ABUF_BYTES), wv);
      __builtin_amdgcn_s_wait_tensorcnt(1);   // tile kt landed; prefetch in flight
    } else {
      __builtin_amdgcn_s_wait_tensorcnt(0);
    }
    __syncthreads();
    mma_step_direct(cur, xb, kt, wm, wn, lid, acc);
    __syncthreads();
  }

  // phase 2: adj @ (rt*y), K = NT
  issue_slice(adj, NT, row0, 0, lbase, wv);
  for (int kt = 0; kt < NT / 32; ++kt) {
    const float* cur = ldsA + (kt & 1) * ABUF_ELEMS;
    if (kt + 1 < NT / 32) {
      issue_slice(adj, NT, row0, (kt + 1) * 32,
                  lbase + (unsigned int)(((kt + 1) & 1) * ABUF_BYTES), wv);
      __builtin_amdgcn_s_wait_tensorcnt(1);
    } else {
      __builtin_amdgcn_s_wait_tensorcnt(0);
    }
    __syncthreads();
    mma_step_direct(cur, yb, kt, wm, wn, lid, acc);
    __syncthreads();
  }

  const int hi = lid >> 4, nn = lid & 15;
  const int m0 = row0 + wm * 16;
#pragma unroll
  for (int nt = 0; nt < 4; ++nt) {
    const int n = (wn * 4 + nt) * 16 + nn;
#pragma unroll
    for (int r = 0; r < 8; ++r) {
      const int m = m0 + r + hi * 8;              // C/D layout: VGPR r -> M=r / M=8+r
      const float rm = rs[m];
      float v = (acc[nt][r] + x_f32[(size_t)m * DOUT + n]) * rm;
      v = fmaxf(v, 0.0f);
      out_x[(size_t)m * DOUT + n] = v;
      xnb[swz_idx(m, n)] = (__bf16)(v * rm);
    }
  }
}

// ---------------- y-update ----------------
// y_new = relu( rt[j] * ( y[j] + adj_t@(rt*y) + adj^T@(rs*x_new) ) )

__global__ void __launch_bounds__(128) k_yupdate(const float* __restrict__ adj,
                                                 const float* __restrict__ adj_t,
                                                 const __bf16* __restrict__ yb,
                                                 const __bf16* __restrict__ xnb,
                                                 const float* __restrict__ y_f32,
                                                 const float* __restrict__ rt,
                                                 float* __restrict__ out_y) {
  __shared__ float ldsA[2 * ABUF_ELEMS];
  const int tid = threadIdx.x;
  const int lid = tid & 31;
  const int wv = __builtin_amdgcn_readfirstlane(tid >> 5);
  const int wm = wv >> 1, wn = wv & 1;
  const int row0 = blockIdx.x * 32;               // j block
  const unsigned int lbase = (unsigned int)(size_t)(void*)&ldsA[0];
  v8f acc[4] = {};

  // phase 1: adj_t @ (rt*y), K = NT (direct)
  issue_slice(adj_t, NT, row0, 0, lbase, wv);
  for (int kt = 0; kt < NT / 32; ++kt) {
    const float* cur = ldsA + (kt & 1) * ABUF_ELEMS;
    if (kt + 1 < NT / 32) {
      issue_slice(adj_t, NT, row0, (kt + 1) * 32,
                  lbase + (unsigned int)(((kt + 1) & 1) * ABUF_BYTES), wv);
      __builtin_amdgcn_s_wait_tensorcnt(1);
    } else {
      __builtin_amdgcn_s_wait_tensorcnt(0);
    }
    __syncthreads();
    mma_step_direct(cur, yb, kt, wm, wn, lid, acc);
    __syncthreads();
  }

  // phase 2: adj^T @ (rs*x_new), K = NS. DMA the 32(i)x32(j) tile un-transposed
  // (rows i of adj, our 32 j columns); transpose happens in the fragment gather.
  issue_slice(adj + row0, NT, 0, 0, lbase, wv);   // rows i=wv*8.., cols j=row0..row0+31
  for (int kt = 0; kt < NS / 32; ++kt) {
    const float* cur = ldsA + (kt & 1) * ABUF_ELEMS;
    if (kt + 1 < NS / 32) {
      issue_slice(adj + row0, NT, (kt + 1) * 32, 0,
                  lbase + (unsigned int)(((kt + 1) & 1) * ABUF_BYTES), wv);
      __builtin_amdgcn_s_wait_tensorcnt(1);
    } else {
      __builtin_amdgcn_s_wait_tensorcnt(0);
    }
    __syncthreads();
    mma_step_trans(cur, xnb, kt, wm, wn, lid, acc);
    __syncthreads();
  }

  const int hi = lid >> 4, nn = lid & 15;
  const int j0 = row0 + wm * 16;
#pragma unroll
  for (int nt = 0; nt < 4; ++nt) {
    const int n = (wn * 4 + nt) * 16 + nn;
#pragma unroll
    for (int r = 0; r < 8; ++r) {
      const int j = j0 + r + hi * 8;
      float v = (acc[nt][r] + y_f32[(size_t)j * DOUT + n]) * rt[j];
      out_y[(size_t)j * DOUT + n] = fmaxf(v, 0.0f);
    }
  }
}

// ---------------- launch ----------------

extern "C" void kernel_launch(void* const* d_in, const int* in_sizes, int n_in,
                              void* d_out, int out_size, void* d_ws, size_t ws_size,
                              hipStream_t stream) {
  const float* inp_s = (const float*)d_in[0];
  const float* inp_t = (const float*)d_in[1];
  const float* adj   = (const float*)d_in[2];
  const float* adj_s = (const float*)d_in[3];
  const float* adj_t = (const float*)d_in[4];
  const float* W     = (const float*)d_in[5];

  char* ws = (char*)d_ws;
  // ws layout (bytes), all offsets 256B-aligned; ~10.1 MB total
  float*  x_f32 = (float*)(ws + 0);                    // NS*128*4 = 2 MB
  float*  y_f32 = (float*)(ws + 2097152);              // NT*128*4 = 4 MB
  float*  rs    = (float*)(ws + 6291456);              // NS*4
  float*  rt    = (float*)(ws + 6307840);              // NT*4
  float*  deg   = (float*)(ws + 6340608);              // NT*4
  __bf16* xb    = (__bf16*)(ws + 6373376);             // NS*128*2 = 1 MB
  __bf16* yb    = (__bf16*)(ws + 7421952);             // NT*128*2 = 2 MB
  __bf16* xnb   = (__bf16*)(ws + 9519104);             // NS*128*2 = 1 MB

  float* out_x = (float*)d_out;
  float* out_y = out_x + (size_t)NS * DOUT;

  k_deg_init<<<NT / 256, 256, 0, stream>>>(deg);
  k_row_deg<<<NS, 256, 0, stream>>>(adj, adj_s, rs);
  dim3 g2(NT / 256, 8);
  k_col_deg<<<g2, 256, 0, stream>>>(adj, adj_t, deg);
  k_rt<<<NT / 256, 256, 0, stream>>>(deg, rt);
  k_proj<<<(NS + NT) / 8, 256, 0, stream>>>(inp_s, inp_t, W, rs, rt, x_f32, y_f32, xb, yb);
  k_xupdate<<<NS / 32, 128, 0, stream>>>(adj, adj_s, xb, yb, x_f32, rs, out_x, xnb);
  k_yupdate<<<NT / 32, 128, 0, stream>>>(adj, adj_t, yb, xnb, y_f32, rt, out_y);
}
